// CrossAttention_31834297598397
// MI455X (gfx1250) — matrix operand order repaired and checked by reading, not compiled, not run on hardware
//
#include <hip/hip_runtime.h>
#include <stdint.h>

typedef __bf16 bf16;
typedef uint32_t u32;
typedef uint64_t u64;
typedef __attribute__((ext_vector_type(8)))  float v8f;
typedef __attribute__((ext_vector_type(16))) bf16  v16bf;
typedef __attribute__((ext_vector_type(8)))  bf16  bf16x8;
typedef __attribute__((ext_vector_type(4)))  float f32x4;
typedef __attribute__((ext_vector_type(4)))  u32   u32x4;
typedef __attribute__((ext_vector_type(8)))  u32   u32x8;

// ---------------------------------------------------------------------------
// WMMA helpers (CDNA5: D = A(16x32 bf16) x B(32x16 bf16) + C(16x16 f32))
// ---------------------------------------------------------------------------
__device__ __forceinline__ v8f wmma_bf16(v16bf a, v16bf b, v8f c) {
  return __builtin_amdgcn_wmma_f32_16x16x32_bf16(false, a, false, b, (short)0, c,
                                                 false, false);
}

__device__ __forceinline__ v16bf frag2(const bf16* p0, const bf16* p1) {
  bf16x8 lo = *(const bf16x8*)p0;
  bf16x8 hi = *(const bf16x8*)p1;
  v16bf r;
#pragma unroll
  for (int i = 0; i < 8; ++i) { r[i] = lo[i]; r[i + 8] = hi[i]; }
  return r;
}

__device__ __forceinline__ v16bf frag2f(const float* p0, const float* p1) {
  v16bf r;
#pragma unroll
  for (int i = 0; i < 8; ++i) { r[i] = (bf16)p0[i]; r[i + 8] = (bf16)p1[i]; }
  return r;
}

__device__ __forceinline__ v8f zero8() {
  v8f z = {0.f, 0.f, 0.f, 0.f, 0.f, 0.f, 0.f, 0.f};
  return z;
}

// ---------------------------------------------------------------------------
// Tensor Data Mover: 2-D bf16 tile (tile_rows x 32 elems) global -> LDS.
// LDS padding: +4 DWORDs after every 16 DWORDs (row 64B -> pitch 80B = 40 hw),
// the bank-conflict-free pitch the compute fragments expect.
// ---------------------------------------------------------------------------
__device__ __forceinline__ void tdm_load_tile32(u32 lds_off, const bf16* gsrc,
                                                u32 tile_rows, u32 tensor_rows,
                                                u32 row_stride_elems) {
  const u64 ga = (u64)(uintptr_t)gsrc;
  u32x4 g0;
  g0[0] = 1u;                                            // count=1 (valid D#)
  g0[1] = lds_off;                                       // lds_addr (bytes)
  g0[2] = (u32)ga;                                       // global_addr[31:0]
  g0[3] = (u32)((ga >> 32) & 0x01FFFFFFu) | 0x80000000u; // addr[56:32]|type=2
  u32x8 g1;
  // data_size=2B (1<<16), pad_enable (1<<20), pad_interval=16DW (3<<22),
  // pad_amount=4DW (3<<25); workgroup_mask=0 (not in cluster)
  g1[0] = (1u << 16) | (1u << 20) | (3u << 22) | (3u << 25);
  g1[1] = (row_stride_elems & 0xFFFFu) << 16;            // tensor_dim0 lo16
  g1[2] = ((row_stride_elems >> 16) & 0xFFFFu) |         // tensor_dim0 hi16
          ((tensor_rows & 0xFFFFu) << 16);               // tensor_dim1 lo16
  g1[3] = ((tensor_rows >> 16) & 0xFFFFu) |              // tensor_dim1 hi16
          (32u << 16);                                   // tile_dim0 = 32
  g1[4] = tile_rows & 0xFFFFu;                           // tile_dim1
  g1[5] = row_stride_elems;                              // dim0_stride lo32
  g1[6] = 0u;
  g1[7] = 0u;
  asm volatile("tensor_load_to_lds %0, %1" ::"s"(g0), "s"(g1) : "memory");
}

// ---------------------------------------------------------------------------
// Kernel: out[M,N] = A[M,K](bf16) @ Bt[N,K](bf16)^T + bias, f32 out.
// Block tile 128x128, 256 threads (8 waves), K-step 32.
// Double-buffered TDM staging: issue next tile's DMA, s_wait_tensorcnt(2) to
// retire only the current tile (TENSORcnt is in-order), compute, swap.
// ---------------------------------------------------------------------------
__global__ __launch_bounds__(256) void gemm_bias_tdm(
    const bf16* __restrict__ A, const bf16* __restrict__ Bt,
    const float* __restrict__ bias, float* __restrict__ out,
    int M, int N, int K) {
  const int LDT = 40;  // halfwords/row (TDM pads 64B rows to 80B pitch)
  __shared__ __align__(128) bf16 As[2][128 * 40];
  __shared__ __align__(128) bf16 Bs[2][128 * 40];

  const int tid = threadIdx.x;
  const int wave = tid >> 5, lane = tid & 31;
  const int half = lane >> 4, lm = lane & 15;
  const int m0g = blockIdx.x * 128, n0g = blockIdx.y * 128;

  const u32 ldsA0 = (u32)(uintptr_t)(void*)&As[0][0];
  const u32 ldsB0 = (u32)(uintptr_t)(void*)&Bs[0][0];
  const u32 bufStride = 128u * 40u * 2u;  // bytes per buffer

  const bf16* Ab = A + (size_t)m0g * K;
  const bf16* Bb = Bt + (size_t)n0g * K;

  v8f acc[8];
#pragma unroll
  for (int i = 0; i < 8; ++i) acc[i] = zero8();

  // Prologue: DMA first K-tile into buffer 0
  if (wave == 0) {
    tdm_load_tile32(ldsA0, Ab, 128u, (u32)(M - m0g), (u32)K);
    tdm_load_tile32(ldsB0, Bb, 128u, (u32)(N - n0g), (u32)K);
  }

  int buf = 0;
  for (int kk = 0; kk < K; kk += 32) {
    if (wave == 0) {
      if (kk + 32 < K) {
        // Prefetch next tile into the other buffer, then wait for current
        // tile only (2 newest outstanding stay in flight).
        tdm_load_tile32(ldsA0 + (u32)(buf ^ 1) * bufStride, Ab + kk + 32, 128u,
                        (u32)(M - m0g), (u32)K);
        tdm_load_tile32(ldsB0 + (u32)(buf ^ 1) * bufStride, Bb + kk + 32, 128u,
                        (u32)(N - n0g), (u32)K);
        __builtin_amdgcn_s_wait_tensorcnt(2);
      } else {
        __builtin_amdgcn_s_wait_tensorcnt(0);
      }
    }
    __syncthreads();  // current buffer ready for all waves

    // A frag: lane -> row 16*wave+lm; 16 contiguous K at 16*half
    const bf16* ap = &As[buf][(16 * wave + lm) * LDT + 16 * half];
    v16bf afr = frag2(ap, ap + 8);
#pragma unroll
    for (int nt = 0; nt < 8; ++nt) {
      // B frag: lane -> col 16*nt+lm; 16 contiguous K at 16*half
      const bf16* bp = &Bs[buf][(16 * nt + lm) * LDT + 16 * half];
      v16bf bfr = frag2(bp, bp + 8);
      acc[nt] = wmma_bf16(afr, bfr, acc[nt]);
    }
    __syncthreads();  // all waves done with this buffer before it is re-DMA'd
    buf ^= 1;
  }

  // Epilogue: C layout -> row = r + 8*half, col = lane&15
#pragma unroll
  for (int nt = 0; nt < 8; ++nt) {
    const int n = n0g + 16 * nt + lm;
    const float bv = bias[n];
#pragma unroll
    for (int r = 0; r < 8; ++r) {
      out[(size_t)(m0g + 16 * wave + r + 8 * half) * N + n] = acc[nt][r] + bv;
    }
  }
}

// ---------------------------------------------------------------------------
// Kernel: elementwise f32 -> bf16 (4 elems/thread)
// ---------------------------------------------------------------------------
__global__ __launch_bounds__(256) void cvt_bf16(const float* __restrict__ in,
                                                bf16* __restrict__ out, int n4) {
  const int i = (blockIdx.x * 256 + threadIdx.x);
  if (i < n4) {
    f32x4 v = *(const f32x4*)(in + (size_t)i * 4);
    bf16* o = out + (size_t)i * 4;
    o[0] = (bf16)v.x; o[1] = (bf16)v.y; o[2] = (bf16)v.z; o[3] = (bf16)v.w;
  }
}

// ---------------------------------------------------------------------------
// Kernel: per-row RMSNorm (eps 1e-6) + residual, f32 -> bf16. Row length 1024.
// ---------------------------------------------------------------------------
__global__ __launch_bounds__(256) void rmsnorm_residual_bf16(
    const float* __restrict__ in, const float* __restrict__ scale,
    bf16* __restrict__ out) {
  __shared__ float red[8];
  const int row = blockIdx.x, tid = threadIdx.x;
  const float* p = in + (size_t)row * 1024 + tid * 4;
  f32x4 v = *(const f32x4*)p;
  float ss = v.x * v.x + v.y * v.y + v.z * v.z + v.w * v.w;
#pragma unroll
  for (int off = 16; off >= 1; off >>= 1) ss += __shfl_xor(ss, off, 32);
  if ((tid & 31) == 0) red[tid >> 5] = ss;
  __syncthreads();
  float tot = 0.f;
#pragma unroll
  for (int i = 0; i < 8; ++i) tot += red[i];
  const float rr = rsqrtf(tot * (1.0f / 1024.0f) + 1e-6f);
  f32x4 s = *(const f32x4*)(scale + tid * 4);
  bf16* o = out + (size_t)row * 1024 + tid * 4;
  o[0] = (bf16)(v.x * rr * s.x + v.x);
  o[1] = (bf16)(v.y * rr * s.y + v.y);
  o[2] = (bf16)(v.z * rr * s.z + v.z);
  o[3] = (bf16)(v.w * rr * s.w + v.w);
}

// ---------------------------------------------------------------------------
// Kernel: transpose + convert: in [Z][R=1024][Cc=1024] f32 -> out [Z][Cc][R]
// bf16. Used for V ([B][S][C] -> [B][C][S]) and weights ([K][N] -> [N][K]).
// ---------------------------------------------------------------------------
__global__ __launch_bounds__(256) void transpose_cvt_bf16(
    const float* __restrict__ in, bf16* __restrict__ out) {
  __shared__ float tile[32][33];
  const int z = blockIdx.z;
  const int c0 = blockIdx.x * 32, r0 = blockIdx.y * 32;
  const int tx = threadIdx.x, ty = threadIdx.y;  // 32 x 8
#pragma unroll
  for (int j = 0; j < 4; ++j) {
    const int r = r0 + ty + j * 8;
    tile[ty + j * 8][tx] = in[((size_t)z * 1024 + r) * 1024 + c0 + tx];
  }
  __syncthreads();
#pragma unroll
  for (int j = 0; j < 4; ++j) {
    const int c = c0 + ty + j * 8;
    out[((size_t)z * 1024 + c) * 1024 + r0 + tx] = (bf16)tile[tx][ty + j * 8];
  }
}

// ---------------------------------------------------------------------------
// Kernel: attention core. One block = one (b,h) and a 64-row q tile.
// scores tile 64x1028 f32 in dynamic LDS. Output bf16; attn_mean via atomics.
// ---------------------------------------------------------------------------
__global__ __launch_bounds__(256) void cross_attention_core(
    const bf16* __restrict__ qb, const bf16* __restrict__ kb,
    const bf16* __restrict__ vt, bf16* __restrict__ yb,
    float* __restrict__ attn_mean) {
  extern __shared__ float sc[];  // [64][1028]
  const int LDSW = 1028;
  const int tid = threadIdx.x;
  const int wave = tid >> 5, lane = tid & 31;
  const int half = lane >> 4, lm = lane & 15;
  const int bh = blockIdx.y, b = bh >> 4, h = bh & 15;
  const int t0 = blockIdx.x * 64;
  const int wr = wave & 3;
  const int wc = wave >> 2;

  // ---- Phase 1: scores = (Q K^T) * D^-1/2 -> LDS ---------------------------
  {
    const bf16* qrow =
        qb + ((size_t)(b * 1024 + t0 + 16 * wr + lm)) * 1024 + h * 64;
    v16bf a0 = frag2(qrow + 16 * half, qrow + 8 + 16 * half);
    v16bf a1 = frag2(qrow + 32 + 16 * half, qrow + 40 + 16 * half);
    for (int ct = 0; ct < 32; ++ct) {
      const int s0 = 512 * wc + 16 * ct;
      const bf16* krow = kb + ((size_t)(b * 1024 + s0 + lm)) * 1024 + h * 64;
      v16bf b0 = frag2(krow + 16 * half, krow + 8 + 16 * half);
      v16bf b1 = frag2(krow + 32 + 16 * half, krow + 40 + 16 * half);
      v8f acc = zero8();
      acc = wmma_bf16(a0, b0, acc);
      acc = wmma_bf16(a1, b1, acc);
#pragma unroll
      for (int r = 0; r < 8; ++r)
        sc[(16 * wr + r + 8 * half) * LDSW + s0 + lm] = acc[r] * 0.125f;
    }
  }
  __syncthreads();

  // ---- Phase 2: per-row softmax + attn_mean --------------------------------
  for (int i = 0; i < 8; ++i) {
    const int rloc = 8 * wave + i;
    float* row = sc + rloc * LDSW;
    float mx = -3.0e38f;
    for (int j = lane; j < 1024; j += 32) mx = fmaxf(mx, row[j]);
#pragma unroll
    for (int off = 16; off >= 1; off >>= 1)
      mx = fmaxf(mx, __shfl_xor(mx, off, 32));
    float sum = 0.f;
    for (int j = lane; j < 1024; j += 32) {
      const float e = __expf(row[j] - mx);
      row[j] = e;
      sum += e;
    }
#pragma unroll
    for (int off = 16; off >= 1; off >>= 1) sum += __shfl_xor(sum, off, 32);
    const float inv = 1.0f / sum;
    float* am = attn_mean + ((size_t)b * 1024 + t0 + rloc) * 1024;
    for (int j = lane; j < 1024; j += 32) {
      const float p = row[j] * inv;
      row[j] = p;
      atomicAdd(am + j, p * (1.0f / 16.0f));
    }
  }
  __syncthreads();

  // ---- Phase 3: y = P @ V --------------------------------------------------
  {
    v8f acc0 = zero8(), acc1 = zero8();
    const float* prow = sc + (16 * wr + lm) * LDSW;
    const bf16* vbase0 =
        vt + ((size_t)(b * 1024 + h * 64 + 32 * wc + lm)) * 1024;
    const bf16* vbase1 = vbase0 + (size_t)16 * 1024;
    for (int ks = 0; ks < 32; ++ks) {
      const int sb = 32 * ks;
      v16bf ap = frag2f(prow + sb + 16 * half, prow + sb + 8 + 16 * half);
      v16bf b0 = frag2(vbase0 + sb + 16 * half, vbase0 + sb + 8 + 16 * half);
      v16bf b1 = frag2(vbase1 + sb + 16 * half, vbase1 + sb + 8 + 16 * half);
      acc0 = wmma_bf16(ap, b0, acc0);
      acc1 = wmma_bf16(ap, b1, acc1);
    }
#pragma unroll
    for (int r = 0; r < 8; ++r) {
      const int t = t0 + 16 * wr + r + 8 * half;
      const size_t o = ((size_t)b * 1024 + t) * 1024 + h * 64 + 32 * wc;
      yb[o + lm] = (bf16)acc0[r];
      yb[o + 16 + lm] = (bf16)acc1[r];
    }
  }
}

// ---------------------------------------------------------------------------
__global__ __launch_bounds__(256) void zero_f32(float* __restrict__ p, int n) {
  const int i = blockIdx.x * blockDim.x + threadIdx.x;
  if (i < n) p[i] = 0.f;
}

// ---------------------------------------------------------------------------
extern "C" void kernel_launch(void* const* d_in, const int* in_sizes, int n_in,
                              void* d_out, int out_size, void* d_ws,
                              size_t ws_size, hipStream_t stream) {
  (void)in_sizes; (void)n_in; (void)out_size; (void)ws_size;
  const float* x   = (const float*)d_in[0];
  const float* enc = (const float*)d_in[1];
  const float* Wq  = (const float*)d_in[2];
  const float* bq  = (const float*)d_in[3];
  const float* Wk  = (const float*)d_in[4];
  const float* bk  = (const float*)d_in[5];
  const float* Wv  = (const float*)d_in[6];
  const float* bv  = (const float*)d_in[7];
  const float* qs  = (const float*)d_in[8];
  const float* ksc = (const float*)d_in[9];
  const float* Wp  = (const float*)d_in[10];
  const float* bp  = (const float*)d_in[11];
  float* out = (float*)d_out;

  const size_t E = 4ull * 1024 * 1024;  // B*T*C elements
  const size_t W1 = 1024ull * 1024;     // one weight matrix
  float* qf = (float*)d_ws;      // q projection f32
  float* kf = qf + E;            // k projection f32
  float* vf = kf + E;            // v projection f32
  bf16* xb  = (bf16*)(vf + E);   // x bf16
  bf16* eb  = xb + E;            // enc bf16
  bf16* qbp = eb + E;            // normed q bf16
  bf16* kbp = qbp + E;           // normed k bf16
  bf16* vtp = kbp + E;           // v transposed bf16 [B][C][S]
  bf16* ybp = vtp + E;           // attention out bf16
  bf16* wqt = ybp + E;           // Wq^T bf16 [N][K]
  bf16* wkt = wqt + W1;
  bf16* wvt = wkt + W1;
  bf16* wpt = wvt + W1;
  float* attn_mean = out + E;

  zero_f32<<<dim3(16384), dim3(256), 0, stream>>>(attn_mean, (int)E);

  // One-time precision conversion / weight transposition
  cvt_bf16<<<dim3(4096), dim3(256), 0, stream>>>(x,   xb, (int)(E / 4));
  cvt_bf16<<<dim3(4096), dim3(256), 0, stream>>>(enc, eb, (int)(E / 4));
  const dim3 tg(32, 32, 1), tb(32, 8);
  transpose_cvt_bf16<<<tg, tb, 0, stream>>>(Wq, wqt);
  transpose_cvt_bf16<<<tg, tb, 0, stream>>>(Wk, wkt);
  transpose_cvt_bf16<<<tg, tb, 0, stream>>>(Wv, wvt);
  transpose_cvt_bf16<<<tg, tb, 0, stream>>>(Wp, wpt);

  // Projections (double-buffered TDM bf16 WMMA GEMMs, 128x128 tiles)
  const dim3 gg(32, 8), gb(256);
  gemm_bias_tdm<<<gg, gb, 0, stream>>>(xb, wqt, bq, qf, 4096, 1024, 1024);
  gemm_bias_tdm<<<gg, gb, 0, stream>>>(eb, wkt, bk, kf, 4096, 1024, 1024);
  gemm_bias_tdm<<<gg, gb, 0, stream>>>(eb, wvt, bv, vf, 4096, 1024, 1024);

  rmsnorm_residual_bf16<<<dim3(4096), dim3(256), 0, stream>>>(qf, qs,  qbp);
  rmsnorm_residual_bf16<<<dim3(4096), dim3(256), 0, stream>>>(kf, ksc, kbp);
  transpose_cvt_bf16<<<dim3(32, 32, 4), tb, 0, stream>>>(vf, vtp);

  const size_t shm = 64 * 1028 * sizeof(float);  // ~257 KB of 320 KB WGP LDS
  cross_attention_core<<<dim3(16, 64), dim3(256), shm, stream>>>(
      qbp, kbp, vtp, ybp, attn_mean);

  gemm_bias_tdm<<<gg, gb, 0, stream>>>(ybp, wpt, bp, out, 4096, 1024, 1024);
}
